// TopKBaseLoss_80281528697452
// MI455X (gfx1250) — compile-verified, matrix operand order checked by reading, not run on hardware
//
#include <hip/hip_runtime.h>

// TopK rank chain for MI455X (gfx1250, wave32).
//
// rank(row, t) = #{j : p[j] > p[t]} + #{j < t : p[j] == p[t]}
// computed as indicator-matrix x ones on the WMMA pipe:
//   A[16x32] = f16 indicators (16-row tile, 32-col chunk per wave)
//   B[32x16] = ones  ->  D[16x16] f32 accumulates per-row counts.
//
// Memory bound: 196 MB streamed once @ 23.3 TB/s ~ 8.4 us floor; zero reuse
// -> non-temporal global_load_b128 (don't churn the 192 MB L2).
// VALU (compare+pack ~80 instr / 2KB chunk) is near-balanced with HBM, so
// occupancy is the lever: keep the K loop ROLLED (#pragma unroll 1) so the
// live set stays ~60 VGPRs, no spills, 8+ waves/SIMD hide NT-load latency.

typedef __attribute__((ext_vector_type(16))) _Float16 v16h;
typedef __attribute__((ext_vector_type(8)))  float    v8f;
typedef __attribute__((ext_vector_type(4)))  float    v4f;

#define BATCH 16384
#define NCLS  1000
#define IGNORE_INDEX (-100)

union APack { unsigned int u[8]; v16h h; };

__device__ __forceinline__ unsigned pk2(bool a, bool b) {
  return (a ? 0x00003C00u : 0u) | (b ? 0x3C000000u : 0u);  // two packed f16 1.0/0.0
}

// One chained-rank stage over matrix x for this wave's 16-row tile.
// Returns the new target (rank) for row m; every lane gets its row's value.
__device__ __forceinline__ int rank_stage(const float* __restrict__ x,
                                          int row, int t,
                                          int lane, int m, int khalf,
                                          v16h bOnes) {
  const bool  found = (t >= 0) && (t < NCLS);   // always true in practice
  const int   tc    = found ? t : 0;
  const float* __restrict__ rowp = x + (long)row * NCLS;
  const float pivot = rowp[tc];                 // p[row][t]

  v8f acc = {0.f, 0.f, 0.f, 0.f, 0.f, 0.f, 0.f, 0.f};

  // 31 full 32-wide chunks: cols [0, 992). Keep ROLLED: occupancy > unroll.
#pragma unroll 1
  for (int k0 = 0; k0 < 992; k0 += 32) {
    const int col = k0 + khalf;
    const int lim = tc - col;                   // e < lim  <=>  (col+e) < tc
    const v4f* p4 = (const v4f*)(rowp + col);
    v4f f0 = __builtin_nontemporal_load(p4 + 0);
    v4f f1 = __builtin_nontemporal_load(p4 + 1);
    v4f f2 = __builtin_nontemporal_load(p4 + 2);
    v4f f3 = __builtin_nontemporal_load(p4 + 3);
    __builtin_prefetch(rowp + col + 32, 0, 3);  // global_prefetch_b8, next chunk

    APack a;
    a.u[0] = pk2((f0[0] > pivot) || ((f0[0] == pivot) && (0  < lim)),
                 (f0[1] > pivot) || ((f0[1] == pivot) && (1  < lim)));
    a.u[1] = pk2((f0[2] > pivot) || ((f0[2] == pivot) && (2  < lim)),
                 (f0[3] > pivot) || ((f0[3] == pivot) && (3  < lim)));
    a.u[2] = pk2((f1[0] > pivot) || ((f1[0] == pivot) && (4  < lim)),
                 (f1[1] > pivot) || ((f1[1] == pivot) && (5  < lim)));
    a.u[3] = pk2((f1[2] > pivot) || ((f1[2] == pivot) && (6  < lim)),
                 (f1[3] > pivot) || ((f1[3] == pivot) && (7  < lim)));
    a.u[4] = pk2((f2[0] > pivot) || ((f2[0] == pivot) && (8  < lim)),
                 (f2[1] > pivot) || ((f2[1] == pivot) && (9  < lim)));
    a.u[5] = pk2((f2[2] > pivot) || ((f2[2] == pivot) && (10 < lim)),
                 (f2[3] > pivot) || ((f2[3] == pivot) && (11 < lim)));
    a.u[6] = pk2((f3[0] > pivot) || ((f3[0] == pivot) && (12 < lim)),
                 (f3[1] > pivot) || ((f3[1] == pivot) && (13 < lim)));
    a.u[7] = pk2((f3[2] > pivot) || ((f3[2] == pivot) && (14 < lim)),
                 (f3[3] > pivot) || ((f3[3] == pivot) && (15 < lim)));

    // D += A(16x32 indicators) x B(32x16 ones): per-row chunk count in f32
    acc = __builtin_amdgcn_wmma_f32_16x16x32_f16(
        /*neg_a=*/false, a.h, /*neg_b=*/false, bOnes,
        /*c_mod=*/(short)0, acc, /*reuse_a=*/false, /*reuse_b=*/false);
  }

  // Tail: cols 992..999 (8 valid), counted scalar by lanes 0..15 (row = m)
  int tailc = 0;
  if (lane < 16) {
    const float* tp = rowp + 992;
#pragma unroll
    for (int jj = 0; jj < 8; ++jj) {
      float v = tp[jj];
      int   j = 992 + jj;
      tailc += ((v > pivot) || ((v == pivot) && (j < tc))) ? 1 : 0;
    }
  }

  // Extract per-row counts from the C/D layout:
  //   row r (r<8) -> acc[r] on lanes 0..15;  row r+8 -> acc[r] on lanes 16..31
  float cnt = 0.f;
#pragma unroll
  for (int r = 0; r < 8; ++r) {
    float lo = __shfl(acc[r], 0, 32);
    float hi = __shfl(acc[r], 16, 32);
    cnt = (m == r)     ? lo : cnt;
    cnt = (m == r + 8) ? hi : cnt;
  }
  int tailAll = __shfl(tailc, m, 32);     // lanes 16..31 fetch their row's tail

  int rank = (int)cnt + tailAll;
  return found ? rank : IGNORE_INDEX;     // jnp.where(found, rank, IGNORE)
}

__global__ __launch_bounds__(256) void topk_rank_chain_kernel(
    const float* __restrict__ x0, const float* __restrict__ x1,
    const float* __restrict__ x2, const int* __restrict__ target,
    int* __restrict__ out)
{
  const int lane  = threadIdx.x & 31;
  const int wave  = threadIdx.x >> 5;
  const int m     = lane & 15;          // row within the 16-row tile
  const int khalf = (lane >> 4) << 4;   // 0 or 16: this lane's K offset in a chunk
  const int row   = (blockIdx.x * 8 + wave) * 16 + m;

  // B = all-ones 16-bit matrix (value-uniform, so its VGPR layout is irrelevant)
  v16h bOnes;
#pragma unroll
  for (int i = 0; i < 16; ++i) bOnes[i] = (_Float16)1.0f;

  int t = target[row];
  if (lane < 16) out[row] = t;          // output[0] = target passthrough

  t = rank_stage(x0, row, t, lane, m, khalf, bOnes);
  if (lane < 16) out[1 * BATCH + row] = t;

  t = rank_stage(x1, row, t, lane, m, khalf, bOnes);
  if (lane < 16) out[2 * BATCH + row] = t;

  t = rank_stage(x2, row, t, lane, m, khalf, bOnes);
  if (lane < 16) out[3 * BATCH + row] = t;
}

extern "C" void kernel_launch(void* const* d_in, const int* in_sizes, int n_in,
                              void* d_out, int out_size, void* d_ws, size_t ws_size,
                              hipStream_t stream) {
  const float* x0     = (const float*)d_in[0];
  const float* x1     = (const float*)d_in[1];
  const float* x2     = (const float*)d_in[2];
  // d_in[3] (x3) is unused by the reference's outputs.
  const int*   target = (const int*)d_in[4];
  int*         out    = (int*)d_out;

  // 16 rows/wave, 8 waves/block -> 128 rows/block -> 128 blocks for 16384 rows
  dim3 grid(BATCH / 128);
  dim3 block(256);
  topk_rank_chain_kernel<<<grid, block, 0, stream>>>(x0, x1, x2, target, out);
}
